// Attention_3736621547996
// MI455X (gfx1250) — matrix operand order repaired and checked by reading, not compiled, not run on hardware
//
#include <hip/hip_runtime.h>
#include <hip/hip_bf16.h>

#define C_DIM  768
#define NHEAD  12
#define HDIM   64
#define NBATCH 8
#define SEQ    1024
#define NTOK   8192
#define QKV_N  2304

#define BM 128
#define BN 128
#define BK 64

typedef __bf16          v16bf __attribute__((ext_vector_type(16)));
typedef float           v8f   __attribute__((ext_vector_type(8)));
typedef float           f32x4 __attribute__((ext_vector_type(4)));
typedef unsigned short  u16x8 __attribute__((ext_vector_type(8)));
typedef unsigned short  u16x4 __attribute__((ext_vector_type(4)));
typedef int             v4i   __attribute__((vector_size(4 * sizeof(int))));

struct BFrag { u16x8 lo, hi; };

static __device__ __forceinline__ v16bf frag_cast(BFrag f) {
    return __builtin_bit_cast(v16bf, f);
}

static __device__ __forceinline__ unsigned short f32_bf16(float f) {
    unsigned int u = __float_as_uint(f);
    u += 0x7FFFu + ((u >> 16) & 1u);     // round-to-nearest-even
    return (unsigned short)(u >> 16);
}

// ---- CDNA5 async global->LDS copy (ASYNCcnt) with safe fallback ------------
#if defined(__HIP_DEVICE_COMPILE__) && defined(__has_builtin)
#if __has_builtin(__builtin_amdgcn_global_load_async_to_lds_b128) && \
    __has_builtin(__builtin_amdgcn_s_wait_asynccnt)
#define HAVE_ASYNC_LDS 1
#endif
#endif
#ifndef HAVE_ASYNC_LDS
#define HAVE_ASYNC_LDS 0
#endif

static __device__ __forceinline__ void async_copy_b128(
    const unsigned short* __restrict__ g, unsigned short* __restrict__ l) {
#if HAVE_ASYNC_LDS
    __builtin_amdgcn_global_load_async_to_lds_b128(
        (__attribute__((address_space(1))) v4i*)g,
        (__attribute__((address_space(3))) v4i*)l, 0, 0);
#else
    *(u16x8*)l = *(const u16x8*)g;
#endif
}

static __device__ __forceinline__ void wait_async_lds() {
#if HAVE_ASYNC_LDS
    __builtin_amdgcn_s_wait_asynccnt(0);
#endif
}

// ---------------------------------------------------------------------------
// One-shot: transpose fp32 weight (K x N) -> bf16 (N x K, row-major).
// ---------------------------------------------------------------------------
__global__ __launch_bounds__(256) void transpose_cast(
    const float* __restrict__ w, unsigned short* __restrict__ wt, int K, int N)
{
    int id = blockIdx.x * 256 + threadIdx.x;
    if (id >= K * N) return;
    int n = id / K;
    int k = id - n * K;
    wt[id] = f32_bf16(w[(size_t)k * N + n]);
}

// ---------------------------------------------------------------------------
// One-shot: elementwise fp32 -> bf16 (x tensor), vectorized 4-wide.
// ---------------------------------------------------------------------------
__global__ __launch_bounds__(256) void cast_bf16(
    const float* __restrict__ in, unsigned short* __restrict__ out, int n4)
{
    int id = blockIdx.x * 256 + threadIdx.x;
    if (id >= n4) return;
    f32x4 v = *(const f32x4*)(in + (size_t)id * 4);
    u16x4 h;
    h[0] = f32_bf16(v[0]); h[1] = f32_bf16(v[1]);
    h[2] = f32_bf16(v[2]); h[3] = f32_bf16(v[3]);
    *(u16x4*)(out + (size_t)id * 4) = h;
}

// ---------------------------------------------------------------------------
// QKV GEMM: (8192 x 768 bf16) x (768 x 2304) + bias -> bf16 Q/K/V buffers
// laid out (B, nh, N, hd). Q pre-scaled by hd^-0.5 = 0.125.
// 128x128 block tile, K-chunk 64, double-buffered async LDS staging.
// ---------------------------------------------------------------------------
__global__ __launch_bounds__(256) void qkv_gemm(
    const unsigned short* __restrict__ X, const unsigned short* __restrict__ Wt,
    const float* __restrict__ bias,
    unsigned short* __restrict__ qbuf, unsigned short* __restrict__ kbuf,
    unsigned short* __restrict__ vbuf)
{
    __shared__ __align__(16) unsigned short As[2][BM * BK];
    __shared__ __align__(16) unsigned short Bs[2][BN * BK];

    const int tid  = threadIdx.x;
    const int wid  = tid >> 5;
    const int lane = tid & 31;
    const int col  = lane & 15;
    const int hi   = lane >> 4;
    const int wm   = wid & 1;
    const int wn   = wid >> 1;
    const int Mbase = blockIdx.y * BM;
    const int Nbase = blockIdx.x * BN;

    auto load_tiles = [&](int buf, int kc) {
        #pragma unroll
        for (int i = 0; i < 4; ++i) {
            int li = tid + i * 256;          // 1024 u16x8 slots per operand
            int row = li >> 3, c8 = li & 7;
            async_copy_b128(X  + (size_t)(Mbase + row) * C_DIM + kc + c8 * 8,
                            &As[buf][row * BK + c8 * 8]);
            async_copy_b128(Wt + (size_t)(Nbase + row) * C_DIM + kc + c8 * 8,
                            &Bs[buf][row * BK + c8 * 8]);
        }
    };

    v8f acc[4][2];
    #pragma unroll
    for (int mt = 0; mt < 4; ++mt)
        #pragma unroll
        for (int nt = 0; nt < 2; ++nt)
            #pragma unroll
            for (int e = 0; e < 8; ++e) acc[mt][nt][e] = 0.0f;

    load_tiles(0, 0);
    wait_async_lds();
    __syncthreads();

    int cur = 0;
    for (int kc = 0; kc < C_DIM; kc += BK) {
        if (kc + BK < C_DIM) load_tiles(cur ^ 1, kc + BK);

        #pragma unroll
        for (int ko = 0; ko < BK; ko += 32) {
            v16bf a[4], b[2];
            #pragma unroll
            for (int mt = 0; mt < 4; ++mt) {
                int row = wm * 64 + mt * 16 + col;
                BFrag f;
                f.lo = *(const u16x8*)&As[cur][row * BK + ko + hi * 8];
                f.hi = *(const u16x8*)&As[cur][row * BK + ko + 16 + hi * 8];
                a[mt] = frag_cast(f);
            }
            #pragma unroll
            for (int nt = 0; nt < 2; ++nt) {
                int row = wn * 32 + nt * 16 + col;
                BFrag f;
                f.lo = *(const u16x8*)&Bs[cur][row * BK + ko + hi * 8];
                f.hi = *(const u16x8*)&Bs[cur][row * BK + ko + 16 + hi * 8];
                b[nt] = frag_cast(f);
            }
            #pragma unroll
            for (int mt = 0; mt < 4; ++mt)
                #pragma unroll
                for (int nt = 0; nt < 2; ++nt)
                    acc[mt][nt] = __builtin_amdgcn_wmma_f32_16x16x32_bf16(
                        false, a[mt], false, b[nt], (short)0, acc[mt][nt], false, false);
        }

        wait_async_lds();
        __syncthreads();
        cur ^= 1;
    }

    #pragma unroll
    for (int mt = 0; mt < 4; ++mt) {
        int gmBase = Mbase + wm * 64 + mt * 16 + hi * 8;   // C-layout: M = e + 8*hi
        #pragma unroll
        for (int nt = 0; nt < 2; ++nt) {
            int gn    = Nbase + wn * 32 + nt * 16 + col;
            int which = gn / C_DIM;
            int rem   = gn - which * C_DIM;
            int head  = rem >> 6;
            int hd    = rem & 63;
            unsigned short* dst = (which == 0) ? qbuf : ((which == 1) ? kbuf : vbuf);
            float bv  = bias[gn];
            float scl = (which == 0) ? 0.125f : 1.0f;
            #pragma unroll
            for (int e = 0; e < 8; ++e) {
                int gm = gmBase + e;
                int bb = gm >> 10, n = gm & 1023;
                float v = (acc[mt][nt][e] + bv) * scl;
                dst[(size_t)(((bb * NHEAD + head) << 10) + n) * HDIM + hd] = f32_bf16(v);
            }
        }
    }
}

// ---------------------------------------------------------------------------
// Flash attention: one block = 64 query rows of one (batch, head).
// 4 waves x 16 rows; stream 64-key blocks; online softmax; O += P*V.
// ---------------------------------------------------------------------------
__global__ __launch_bounds__(128) void attention(
    const unsigned short* __restrict__ qbuf,
    const unsigned short* __restrict__ kbuf,
    const unsigned short* __restrict__ vbuf,
    unsigned short* __restrict__ attn_out)
{
    __shared__ __align__(16) unsigned short Qs[64 * 64];
    __shared__ __align__(16) unsigned short Ks[64 * 64];
    __shared__ __align__(16) unsigned short Vts[64 * 64];     // V transposed [d][key]
    __shared__ __align__(16) unsigned short Ps[4][16 * 64];   // per-wave P

    const int tid  = threadIdx.x;
    const int wid  = tid >> 5;
    const int lane = tid & 31;
    const int col  = lane & 15;
    const int hi   = lane >> 4;

    const int bid  = blockIdx.x;
    const int qb   = bid & 15;
    const int head = (bid >> 4) % NHEAD;
    const int b    = bid / (16 * NHEAD);

    const size_t bh = ((size_t)b * NHEAD + head) * SEQ;
    const unsigned short* qp = qbuf + (bh + (size_t)qb * 64) * HDIM;
    const unsigned short* kp = kbuf + bh * HDIM;
    const unsigned short* vp = vbuf + bh * HDIM;

    // Load Q tile (64x64 bf16) asynchronously; consumed after first barrier.
    #pragma unroll
    for (int i = 0; i < 4; ++i) {
        int li = tid + i * 128;
        int row = li >> 3, c8 = li & 7;
        async_copy_b128(qp + (size_t)row * HDIM + c8 * 8, &Qs[row * 64 + c8 * 8]);
    }

    float m_r[8], l_r[8];
    v8f o[4];
    #pragma unroll
    for (int e = 0; e < 8; ++e) { m_r[e] = -1e30f; l_r[e] = 0.0f; }
    #pragma unroll
    for (int dt = 0; dt < 4; ++dt)
        #pragma unroll
        for (int e = 0; e < 8; ++e) o[dt][e] = 0.0f;

    for (int kb = 0; kb < SEQ / 64; ++kb) {
        __syncthreads();   // previous iteration done reading Ks/Vts
        #pragma unroll
        for (int i = 0; i < 4; ++i) {
            int li = tid + i * 128;
            int row = li >> 3, c8 = li & 7;
            async_copy_b128(kp + (size_t)(kb * 64 + row) * HDIM + c8 * 8,
                            &Ks[row * 64 + c8 * 8]);
            u16x8 vv = *(const u16x8*)(vp + (size_t)(kb * 64 + row) * HDIM + c8 * 8);
            #pragma unroll
            for (int j = 0; j < 8; ++j) Vts[(c8 * 8 + j) * 64 + row] = vv[j];
        }
        wait_async_lds();      // covers Q (iter 0) + this K tile
        __syncthreads();

        // S = Q * K^T
        v8f s[4];
        #pragma unroll
        for (int jt = 0; jt < 4; ++jt)
            #pragma unroll
            for (int e = 0; e < 8; ++e) s[jt][e] = 0.0f;

        #pragma unroll
        for (int ko = 0; ko < 64; ko += 32) {
            BFrag fa;
            int qrow = wid * 16 + col;
            fa.lo = *(const u16x8*)&Qs[qrow * 64 + ko + hi * 8];
            fa.hi = *(const u16x8*)&Qs[qrow * 64 + ko + 16 + hi * 8];
            v16bf a = frag_cast(fa);
            #pragma unroll
            for (int jt = 0; jt < 4; ++jt) {
                int j = jt * 16 + col;
                BFrag fb;
                fb.lo = *(const u16x8*)&Ks[j * 64 + ko + hi * 8];
                fb.hi = *(const u16x8*)&Ks[j * 64 + ko + 16 + hi * 8];
                s[jt] = __builtin_amdgcn_wmma_f32_16x16x32_bf16(
                    false, a, false, frag_cast(fb), (short)0, s[jt], false, false);
            }
        }

        // Online softmax (rows live across 16-lane halves; shfl_xor reductions)
        #pragma unroll
        for (int e = 0; e < 8; ++e) {
            float mx = s[0][e];
            #pragma unroll
            for (int jt = 1; jt < 4; ++jt) mx = fmaxf(mx, s[jt][e]);
            #pragma unroll
            for (int msk = 1; msk < 16; msk <<= 1)
                mx = fmaxf(mx, __shfl_xor(mx, msk, 32));
            float mnew = fmaxf(m_r[e], mx);
            float corr = __expf(m_r[e] - mnew);
            float sum = 0.0f;
            #pragma unroll
            for (int jt = 0; jt < 4; ++jt) {
                float p = __expf(s[jt][e] - mnew);
                s[jt][e] = p;
                sum += p;
            }
            #pragma unroll
            for (int msk = 1; msk < 16; msk <<= 1)
                sum += __shfl_xor(sum, msk, 32);
            l_r[e] = l_r[e] * corr + sum;
            m_r[e] = mnew;
            #pragma unroll
            for (int dt = 0; dt < 4; ++dt) o[dt][e] *= corr;
        }

        // C-layout -> A-layout via LDS round trip
        #pragma unroll
        for (int jt = 0; jt < 4; ++jt)
            #pragma unroll
            for (int e = 0; e < 8; ++e)
                Ps[wid][(e + 8 * hi) * 64 + jt * 16 + col] = f32_bf16(s[jt][e]);
        __syncthreads();

        // O += P * V
        #pragma unroll
        for (int ko = 0; ko < 64; ko += 32) {
            BFrag fa;
            fa.lo = *(const u16x8*)&Ps[wid][col * 64 + ko + hi * 8];
            fa.hi = *(const u16x8*)&Ps[wid][col * 64 + ko + 16 + hi * 8];
            v16bf a = frag_cast(fa);
            #pragma unroll
            for (int dt = 0; dt < 4; ++dt) {
                int d = dt * 16 + col;
                BFrag fb;
                fb.lo = *(const u16x8*)&Vts[d * 64 + ko + hi * 8];
                fb.hi = *(const u16x8*)&Vts[d * 64 + ko + 16 + hi * 8];
                o[dt] = __builtin_amdgcn_wmma_f32_16x16x32_bf16(
                    false, a, false, frag_cast(fb), (short)0, o[dt], false, false);
            }
        }
    }

    // Normalize + write bf16 (token-major (T, C) for proj GEMM)
    #pragma unroll
    for (int dt = 0; dt < 4; ++dt) {
        int cdim = head * 64 + dt * 16 + col;
        #pragma unroll
        for (int e = 0; e < 8; ++e) {
            int row = e + 8 * hi;
            int tok = b * SEQ + qb * 64 + wid * 16 + row;
            attn_out[(size_t)tok * C_DIM + cdim] = f32_bf16(o[dt][e] / l_r[e]);
        }
    }
}

// ---------------------------------------------------------------------------
// Proj GEMM: (8192 x 768 bf16) x (768 x 768) + bias -> fp32 output.
// Same double-buffered async structure as qkv_gemm.
// ---------------------------------------------------------------------------
__global__ __launch_bounds__(256) void proj_gemm(
    const unsigned short* __restrict__ A, const unsigned short* __restrict__ Wt,
    const float* __restrict__ bias, float* __restrict__ out)
{
    __shared__ __align__(16) unsigned short As[2][BM * BK];
    __shared__ __align__(16) unsigned short Bs[2][BN * BK];

    const int tid  = threadIdx.x;
    const int wid  = tid >> 5;
    const int lane = tid & 31;
    const int col  = lane & 15;
    const int hi   = lane >> 4;
    const int wm   = wid & 1;
    const int wn   = wid >> 1;
    const int Mbase = blockIdx.y * BM;
    const int Nbase = blockIdx.x * BN;

    auto load_tiles = [&](int buf, int kc) {
        #pragma unroll
        for (int i = 0; i < 4; ++i) {
            int li = tid + i * 256;
            int row = li >> 3, c8 = li & 7;
            async_copy_b128(A  + (size_t)(Mbase + row) * C_DIM + kc + c8 * 8,
                            &As[buf][row * BK + c8 * 8]);
            async_copy_b128(Wt + (size_t)(Nbase + row) * C_DIM + kc + c8 * 8,
                            &Bs[buf][row * BK + c8 * 8]);
        }
    };

    v8f acc[4][2];
    #pragma unroll
    for (int mt = 0; mt < 4; ++mt)
        #pragma unroll
        for (int nt = 0; nt < 2; ++nt)
            #pragma unroll
            for (int e = 0; e < 8; ++e) acc[mt][nt][e] = 0.0f;

    load_tiles(0, 0);
    wait_async_lds();
    __syncthreads();

    int cur = 0;
    for (int kc = 0; kc < C_DIM; kc += BK) {
        if (kc + BK < C_DIM) load_tiles(cur ^ 1, kc + BK);

        #pragma unroll
        for (int ko = 0; ko < BK; ko += 32) {
            v16bf a[4], bfr[2];
            #pragma unroll
            for (int mt = 0; mt < 4; ++mt) {
                int row = wm * 64 + mt * 16 + col;
                BFrag f;
                f.lo = *(const u16x8*)&As[cur][row * BK + ko + hi * 8];
                f.hi = *(const u16x8*)&As[cur][row * BK + ko + 16 + hi * 8];
                a[mt] = frag_cast(f);
            }
            #pragma unroll
            for (int nt = 0; nt < 2; ++nt) {
                int row = wn * 32 + nt * 16 + col;
                BFrag f;
                f.lo = *(const u16x8*)&Bs[cur][row * BK + ko + hi * 8];
                f.hi = *(const u16x8*)&Bs[cur][row * BK + ko + 16 + hi * 8];
                bfr[nt] = frag_cast(f);
            }
            #pragma unroll
            for (int mt = 0; mt < 4; ++mt)
                #pragma unroll
                for (int nt = 0; nt < 2; ++nt)
                    acc[mt][nt] = __builtin_amdgcn_wmma_f32_16x16x32_bf16(
                        false, a[mt], false, bfr[nt], (short)0, acc[mt][nt], false, false);
        }

        wait_async_lds();
        __syncthreads();
        cur ^= 1;
    }

    #pragma unroll
    for (int mt = 0; mt < 4; ++mt) {
        int gmBase = Mbase + wm * 64 + mt * 16 + hi * 8;
        #pragma unroll
        for (int nt = 0; nt < 2; ++nt) {
            int gn = Nbase + wn * 32 + nt * 16 + col;
            float bv = bias[gn];
            #pragma unroll
            for (int e = 0; e < 8; ++e)
                out[(size_t)(gmBase + e) * C_DIM + gn] = acc[mt][nt][e] + bv;
        }
    }
}

// ---------------------------------------------------------------------------
extern "C" void kernel_launch(void* const* d_in, const int* in_sizes, int n_in,
                              void* d_out, int out_size, void* d_ws, size_t ws_size,
                              hipStream_t stream) {
    const float* x      = (const float*)d_in[0];
    const float* w_qkv  = (const float*)d_in[1];
    const float* b_qkv  = (const float*)d_in[2];
    const float* w_proj = (const float*)d_in[3];
    const float* b_proj = (const float*)d_in[4];
    float* out = (float*)d_out;

    char* ws = (char*)d_ws;
    size_t off = 0;
    const size_t qkv_elems = (size_t)NBATCH * NHEAD * SEQ * HDIM;   // 6,291,456
    unsigned short* qbuf   = (unsigned short*)(ws + off); off += qkv_elems * 2;
    unsigned short* kbuf   = (unsigned short*)(ws + off); off += qkv_elems * 2;
    unsigned short* vbuf   = (unsigned short*)(ws + off); off += qkv_elems * 2;
    unsigned short* attn   = (unsigned short*)(ws + off); off += (size_t)NTOK * C_DIM * 2;
    unsigned short* wqkvT  = (unsigned short*)(ws + off); off += (size_t)QKV_N * C_DIM * 2;
    unsigned short* wprojT = (unsigned short*)(ws + off); off += (size_t)C_DIM * C_DIM * 2;
    unsigned short* xbf    = (unsigned short*)(ws + off); off += (size_t)NTOK * C_DIM * 2;

    cast_bf16<<<(NTOK * C_DIM / 4 + 255) / 256, 256, 0, stream>>>(
        x, xbf, NTOK * C_DIM / 4);
    transpose_cast<<<(QKV_N * C_DIM + 255) / 256, 256, 0, stream>>>(w_qkv, wqkvT, C_DIM, QKV_N);
    transpose_cast<<<(C_DIM * C_DIM + 255) / 256, 256, 0, stream>>>(w_proj, wprojT, C_DIM, C_DIM);

    qkv_gemm<<<dim3(QKV_N / BN, NTOK / BM), 256, 0, stream>>>(
        xbf, wqkvT, b_qkv, qbuf, kbuf, vbuf);

    attention<<<NBATCH * NHEAD * (SEQ / 64), 128, 0, stream>>>(
        qbuf, kbuf, vbuf, attn);

    proj_gemm<<<dim3(C_DIM / BN, NTOK / BM), 256, 0, stream>>>(
        attn, wprojT, b_proj, out);
}
